// INT8SQLinear_12360915878784
// MI455X (gfx1250) — 8-bit matrix operand order repaired and checked by reading, compiled, not run on hardware
//
#include <hip/hip_runtime.h>
#include <hip/hip_bf16.h>


typedef __attribute__((ext_vector_type(8))) int       v8i;
typedef __attribute__((ext_vector_type(8))) _Float16  v8h;
typedef _Float16 half_t;

#define M_DIM  8192
#define K_DIM  4096
#define N_DIM  11008
#define TILE_M 128
#define TILE_N 256
#define TILE_K 64
#define NK     (K_DIM / TILE_K)

__device__ __forceinline__ void async_copy_b128(void* lds_dst, const void* g_src)
{
    const unsigned lds_off = (unsigned)(unsigned long long)lds_dst;
    asm volatile("global_load_async_to_lds_b128 %0, %1, off"
                 :: "v"(lds_off), "v"(g_src)
                 : "memory");
}
__device__ __forceinline__ void wait_async_zero()
{
    asm volatile("s_wait_asynccnt 0x0" ::: "memory");
}

__global__ __launch_bounds__(256)
void quant_act_kernel(const half_t* __restrict__ x,
                      const half_t* __restrict__ pqs,
                      const float*  __restrict__ s_ptr,
                      signed char*  __restrict__ qa)
{
    const float inv_s = 1.0f / s_ptr[0];
    const long  base  = ((long)blockIdx.x * blockDim.x + threadIdx.x) * 8;
    const int   col   = (int)(base & (K_DIM - 1));

    v8h xv = *(const v8h*)(x + base);
    v8h pv = *(const v8h*)(pqs + col);

    union { signed char b[8]; int2 v; } u;
#pragma unroll
    for (int e = 0; e < 8; ++e) {
        const half_t sm = xv[e] * pv[e];
        float q = rintf((float)sm * inv_s);
        q = fminf(127.0f, fmaxf(-128.0f, q));
        u.b[e] = (signed char)(int)q;
    }
    *(int2*)(qa + base) = u.v;
}

__global__ __launch_bounds__(256)
void pack_w_kernel(const int* __restrict__ w, int* __restrict__ qw)
{
    const long t = (long)blockIdx.x * blockDim.x + threadIdx.x;
    const int4 v = *(const int4*)(w + t * 4);
    qw[t] = (v.x & 0xFF) | ((v.y & 0xFF) << 8) |
            ((v.z & 0xFF) << 16) | ((v.w & 0xFF) << 24);
}

__global__ __launch_bounds__(256)
void int8_gemm_kernel(const signed char* __restrict__ qa,
                      const signed char* __restrict__ qw,
                      const float*       __restrict__ wscale,
                      const float*       __restrict__ s_ptr,
                      half_t*            __restrict__ out)
{
    __shared__ __align__(16) signed char lA[2][TILE_M * TILE_K];
    __shared__ __align__(16) signed char lB[2][TILE_N * TILE_K];

    const int tid   = threadIdx.x;
    const int lane  = tid & 31;
    const int wv    = tid >> 5;
    const int wm    = (wv >> 2) * 64;
    const int wn    = (wv & 3) * 64;
    const int mBase = blockIdx.y * TILE_M;
    const int nBase = blockIdx.x * TILE_N;

    const int l16   = lane & 15;
    const int lhalf = lane >> 4;
    const int aKoff = lhalf * 16;
    const int bKoff = lhalf * 16;

    const int ldr = tid >> 2;
    const int ldc = (tid & 3) * 16;

    const signed char* gA[2] = {
        qa + (long)(mBase + ldr)      * K_DIM + ldc,
        qa + (long)(mBase + ldr + 64) * K_DIM + ldc };
    const signed char* gB[4] = {
        qw + (long)(nBase + ldr)       * K_DIM + ldc,
        qw + (long)(nBase + ldr +  64) * K_DIM + ldc,
        qw + (long)(nBase + ldr + 128) * K_DIM + ldc,
        qw + (long)(nBase + ldr + 192) * K_DIM + ldc };

    const unsigned sRow = ldr * 64 + ldc;

    unsigned aOff[4], bOff[4];
#pragma unroll
    for (int i = 0; i < 4; ++i)
        aOff[i] = (unsigned)((wm + i * 16 + l16) * 64 + aKoff);
#pragma unroll
    for (int j = 0; j < 4; ++j)
        bOff[j] = (unsigned)((wn + j * 16 + l16) * 64 + bKoff);

    async_copy_b128(&lA[0][sRow],          gA[0]);
    async_copy_b128(&lA[0][sRow + 4096],   gA[1]);
#pragma unroll
    for (int p = 0; p < 4; ++p)
        async_copy_b128(&lB[0][sRow + p * 4096], gB[p]);
    wait_async_zero();
    __syncthreads();

    v8i acc[4][4] = {};

    for (int kt = 0; kt < NK - 1; ++kt) {
        const int cur = kt & 1;
        const int nxt = cur ^ 1;
        const long goff = (long)(kt + 1) * TILE_K;

        async_copy_b128(&lA[nxt][sRow],        gA[0] + goff);
        async_copy_b128(&lA[nxt][sRow + 4096], gA[1] + goff);
#pragma unroll
        for (int p = 0; p < 4; ++p)
            async_copy_b128(&lB[nxt][sRow + p * 4096], gB[p] + goff);

        v8i bfr[4];
#pragma unroll
        for (int j = 0; j < 4; ++j) {
            const signed char* bp = &lB[cur][bOff[j]];
            const int4 lo = *(const int4*)bp;
            const int4 hi = *(const int4*)(bp + 32);
            bfr[j][0] = lo.x; bfr[j][1] = lo.y; bfr[j][2] = lo.z; bfr[j][3] = lo.w;
            bfr[j][4] = hi.x; bfr[j][5] = hi.y; bfr[j][6] = hi.z; bfr[j][7] = hi.w;
        }
        v8i afr[4];
#pragma unroll
        for (int i = 0; i < 4; ++i) {
            const signed char* ap = &lA[cur][aOff[i]];
#pragma unroll
            for (int c = 0; c < 4; ++c) {
                const int2 v = *(const int2*)(ap + (c & 1) * 8 + (c >> 1) * 32);
                afr[i][2 * c] = v.x; afr[i][2 * c + 1] = v.y;
            }
        }

#pragma unroll
        for (int i = 0; i < 4; ++i)
#pragma unroll
            for (int j = 0; j < 4; ++j)
                acc[i][j] = __builtin_amdgcn_wmma_i32_16x16x64_iu8(
                    true, afr[i], true, bfr[j], acc[i][j], false, false);

        wait_async_zero();
        __syncthreads();
    }

    {
        const int cur = (NK - 1) & 1;
        v8i bfr[4];
#pragma unroll
        for (int j = 0; j < 4; ++j) {
            const signed char* bp = &lB[cur][bOff[j]];
            const int4 lo = *(const int4*)bp;
            const int4 hi = *(const int4*)(bp + 32);
            bfr[j][0] = lo.x; bfr[j][1] = lo.y; bfr[j][2] = lo.z; bfr[j][3] = lo.w;
            bfr[j][4] = hi.x; bfr[j][5] = hi.y; bfr[j][6] = hi.z; bfr[j][7] = hi.w;
        }
        v8i afr[4];
#pragma unroll
        for (int i = 0; i < 4; ++i) {
            const signed char* ap = &lA[cur][aOff[i]];
#pragma unroll
            for (int c = 0; c < 4; ++c) {
                const int2 v = *(const int2*)(ap + (c & 1) * 8 + (c >> 1) * 32);
                afr[i][2 * c] = v.x; afr[i][2 * c + 1] = v.y;
            }
        }
#pragma unroll
        for (int i = 0; i < 4; ++i)
#pragma unroll
            for (int j = 0; j < 4; ++j)
                acc[i][j] = __builtin_amdgcn_wmma_i32_16x16x64_iu8(
                    true, afr[i], true, bfr[j], acc[i][j], false, false);
    }

    const float s = s_ptr[0];
#pragma unroll
    for (int j = 0; j < 4; ++j) {
        const int   col = nBase + wn + j * 16 + l16;
        const float f   = s * wscale[col];
#pragma unroll
        for (int i = 0; i < 4; ++i) {
            const int rowBase = mBase + wm + i * 16 + lhalf * 8;
#pragma unroll
            for (int r = 0; r < 8; ++r) {
                out[(long)(rowBase + r) * N_DIM + col] =
                    (half_t)((float)acc[i][j][r] * f);
            }
        }
    }
}

extern "C" void kernel_launch(void* const* d_in, const int* in_sizes, int n_in,
                              void* d_out, int out_size, void* d_ws, size_t ws_size,
                              hipStream_t stream)
{
    (void)in_sizes; (void)n_in; (void)out_size; (void)ws_size;

    const half_t* hs  = (const half_t*)d_in[0];
    const int*    w   = (const int*)   d_in[1];
    const float*  wsc = (const float*) d_in[2];
    const float*  isc = (const float*) d_in[3];
    const half_t* pqs = (const half_t*)d_in[4];
    half_t*       out = (half_t*)d_out;

    signed char* qa = (signed char*)d_ws;
    signed char* qw = qa + (size_t)M_DIM * K_DIM;

    {
        const long threads = (long)M_DIM * K_DIM / 8;
        quant_act_kernel<<<(unsigned)(threads / 256), 256, 0, stream>>>(hs, pqs, isc, qa);
    }
    {
        const long threads = (long)N_DIM * K_DIM / 4;
        pack_w_kernel<<<(unsigned)(threads / 256), 256, 0, stream>>>(w, (int*)qw);
    }
    {
        dim3 grid(N_DIM / TILE_N, M_DIM / TILE_M);
        int8_gemm_kernel<<<grid, 256, 0, stream>>>(qa, qw, wsc, isc, out);
    }
}
